// OneDConvertedParallelMHA_77850577207443
// MI455X (gfx1250) — matrix operand order repaired and checked by reading, not compiled, not run on hardware
//
#include <hip/hip_runtime.h>

typedef __bf16 bf16;
typedef __attribute__((ext_vector_type(16))) __bf16 v16bf;
typedef __attribute__((ext_vector_type(8)))  __bf16 v8bf;
typedef __attribute__((ext_vector_type(8)))  float  v8f;
typedef __attribute__((ext_vector_type(4)))  int    i32x4;

#define AS1 __attribute__((address_space(1)))
#define AS3 __attribute__((address_space(3)))

#define SEQ   2048
#define EMB   2048
#define HEADS 16
#define DHEAD 128
#define BATCH 4
#define MROWS (BATCH*SEQ)          // 8192
#define LN10000 9.210340371976184f
#define SM_SCALE 0.08838834764831845f   // 1/sqrt(128)

#if __has_builtin(__builtin_amdgcn_global_load_async_to_lds_b128)
#define HAVE_ASYNC_LDS 1
#else
#define HAVE_ASYNC_LDS 0
#endif

static __device__ __forceinline__ v8f zero_v8f() {
    v8f z = {0.f,0.f,0.f,0.f,0.f,0.f,0.f,0.f};
    return z;
}
static __device__ __forceinline__ v16bf cat8(v8bf lo, v8bf hi) {
    return __builtin_shufflevector(lo, hi, 0,1,2,3,4,5,6,7,8,9,10,11,12,13,14,15);
}
static __device__ __forceinline__ v8f wmma_bf16(v16bf a, v16bf b, v8f c) {
    return __builtin_amdgcn_wmma_f32_16x16x32_bf16(false, a, false, b, (short)0, c, false, false);
}

// 16-byte global -> LDS copy: async path on gfx1250 (ASYNCcnt),
// falls back to a synchronous VGPR round-trip if the builtin is absent.
static __device__ __forceinline__ void cp16_async(const bf16* __restrict__ g, bf16* l) {
#if HAVE_ASYNC_LDS
    __builtin_amdgcn_global_load_async_to_lds_b128(
        (AS1 i32x4*)(unsigned long long)(size_t)g,
        (AS3 i32x4*)(unsigned)(size_t)l, 0, 0);
#else
    *(v8bf*)l = *(const v8bf*)g;
#endif
}
static __device__ __forceinline__ void wait_async_all() {
#if HAVE_ASYNC_LDS
#if __has_builtin(__builtin_amdgcn_s_wait_asynccnt)
    __builtin_amdgcn_s_wait_asynccnt(0);
#else
    asm volatile("s_wait_asynccnt 0x0" ::: "memory");
#endif
#endif
}

// ---------------------------------------------------------------- fp32 -> bf16
__global__ void cvt_f32_bf16(const float* __restrict__ in, bf16* __restrict__ out, int n) {
    int i = blockIdx.x * blockDim.x + threadIdx.x;
    int stride = gridDim.x * blockDim.x;
    for (; i < n; i += stride) out[i] = (bf16)in[i];
}

// ---------------------------------------------------------------- GEMM  C = A(bf16,[M][K]) * B(bf16,[N][K])^T + bias
// mode 0: QKV projection -> fused deinterleave+RoPE epilogue, writes Qr/Kr [B,H,S,D] and Vt [B,H,D,S] (bf16)
// mode 1: output projection -> fp32 store to Out [M][N]
__global__ __launch_bounds__(256)
void gemm_bf16(const bf16* __restrict__ A, const bf16* __restrict__ Bm,
               const float* __restrict__ bias, int N, int K, int mode,
               bf16* __restrict__ Qr, bf16* __restrict__ Kr, bf16* __restrict__ Vt,
               float* __restrict__ Out)
{
    __shared__ __align__(32) bf16 As[2][128][32];   // double-buffered: 2 x 8KB
    __shared__ __align__(32) bf16 Bs[2][128][32];   // double-buffered: 2 x 8KB

    const int tid  = threadIdx.x;
    const int lane = tid & 31, wave = tid >> 5;
    const int l16  = lane & 15, lhi  = lane >> 4;
    const int mW   = (wave >> 2) * 64;   // 2 waves along M
    const int nW   = (wave & 3) * 32;    // 4 waves along N
    const int mBase = blockIdx.y * 128;
    const int nBase = blockIdx.x * 128;

    auto issue_tile = [&](int k0, int buf) {
#pragma unroll
        for (int p = 0; p < 2; ++p) {
            const int idx = (tid + p * 256) * 8;     // over 128x32 tile
            const int r = idx >> 5, c = idx & 31;
            cp16_async(A  + (size_t)(mBase + r) * K + k0 + c, &As[buf][r][c]);
            cp16_async(Bm + (size_t)(nBase + r) * K + k0 + c, &Bs[buf][r][c]);
        }
    };

    v8f acc[4][2];
#pragma unroll
    for (int mt = 0; mt < 4; ++mt)
#pragma unroll
        for (int nt = 0; nt < 2; ++nt) acc[mt][nt] = zero_v8f();

    issue_tile(0, 0);
    for (int k0 = 0; k0 < K; k0 += 32) {
        const int cur = (k0 >> 5) & 1;
        wait_async_all();          // our tile's async copies done
        __syncthreads();           // everyone's copies done (also waits dscnt)
        if (k0 + 32 < K) issue_tile(k0 + 32, cur ^ 1);   // prefetch next tile

        v16bf bfrag[2];
#pragma unroll
        for (int nt = 0; nt < 2; ++nt)
            bfrag[nt] = *(const v16bf*)&Bs[cur][nW + nt * 16 + l16][lhi * 16];
#pragma unroll
        for (int mt = 0; mt < 4; ++mt) {
            const int r = mW + mt * 16 + l16;
            v8bf lo = *(const v8bf*)&As[cur][r][lhi * 16];
            v8bf hi = *(const v8bf*)&As[cur][r][lhi * 16 + 8];
            v16bf afrag = cat8(lo, hi);
#pragma unroll
            for (int nt = 0; nt < 2; ++nt)
                acc[mt][nt] = wmma_bf16(afrag, bfrag[nt], acc[mt][nt]);
        }
    }

    // ------------- epilogue -------------
#pragma unroll
    for (int mt = 0; mt < 4; ++mt) {
#pragma unroll
        for (int nt = 0; nt < 2; ++nt) {
            const int n = nBase + nW + nt * 16 + l16;   // wave-uniform head/which per tile
#pragma unroll
            for (int i = 0; i < 8; ++i) {
                const int m = mBase + mW + mt * 16 + 8 * lhi + i;
                float val = acc[mt][nt][i] + bias[n];
                if (mode == 1) {
                    Out[(size_t)m * N + n] = val;
                } else {
                    const int h = n / 384, nh = n % 384;
                    const int which = nh >> 7;           // 0=q 1=k 2=v (uniform over wave)
                    const int d = nh & 127;
                    const int b = m >> 11, s = m & (SEQ - 1);
                    if (which == 2) {
                        Vt[(((size_t)(b * HEADS + h)) * DHEAD + d) * SEQ + s] = (bf16)val;
                    } else {
                        float other = __shfl_xor(val, 1, 32);   // partner column (pair)
                        const int pi = d >> 1;
                        float inv = __expf(-LN10000 * (float)(2 * pi) * (1.0f / 128.0f));
                        float ang = (float)s * inv;
                        float sn, cs; __sincosf(ang, &sn, &cs);
                        float o; int dim;
                        if ((d & 1) == 0) { o = val * cs - other * sn; dim = pi; }
                        else              { o = other * sn + val * cs; dim = 64 + pi; }
                        bf16* dst = (which == 0) ? Qr : Kr;
                        dst[(((size_t)(b * HEADS + h)) * SEQ + s) * DHEAD + dim] = (bf16)o;
                    }
                }
            }
        }
    }
}

// ---------------------------------------------------------------- flash attention (causal)
// grid: (S/128, H, B); block 256 = 8 waves; wave w owns 16 query rows.
__global__ __launch_bounds__(256)
void attn_flash(const bf16* __restrict__ Qr, const bf16* __restrict__ Kr,
                const bf16* __restrict__ Vt, bf16* __restrict__ ctx)
{
    __shared__ __align__(32) bf16 Ks[64][128];    // [key][d]      16KB
    __shared__ __align__(32) bf16 Vs[128][64];    // [d][key]      16KB
    __shared__ __align__(32) bf16 Ps[8][16][64];  // P staging     16KB

    const int tid  = threadIdx.x;
    const int lane = tid & 31, wave = tid >> 5;
    const int l16  = lane & 15, lhi  = lane >> 4;
    const int qt = blockIdx.x, h = blockIdx.y, b = blockIdx.z;

    const size_t bh = (size_t)(b * HEADS + h);
    const bf16* Qp = Qr + bh * SEQ * DHEAD;
    const bf16* Kp = Kr + bh * SEQ * DHEAD;
    const bf16* Vp = Vt + bh * DHEAD * SEQ;

    // Q fragments held in registers: 16 rows x 128 d
    const int qrow = qt * 128 + wave * 16 + l16;
    v16bf qf[4];
#pragma unroll
    for (int kk = 0; kk < 4; ++kk) {
        v8bf lo = *(const v8bf*)(Qp + (size_t)qrow * DHEAD + kk * 32 + lhi * 16);
        v8bf hi = *(const v8bf*)(Qp + (size_t)qrow * DHEAD + kk * 32 + lhi * 16 + 8);
        qf[kk] = cat8(lo, hi);
    }

    float mrow[8], lrow[8];
    v8f oacc[8];
#pragma unroll
    for (int i = 0; i < 8; ++i) { mrow[i] = -3.0e38f; lrow[i] = 0.f; }
#pragma unroll
    for (int dt = 0; dt < 8; ++dt) oacc[dt] = zero_v8f();

    const int ktMax = 2 * qt + 1;                 // causal: keys up to qt*128+127
    for (int kt = 0; kt <= ktMax; ++kt) {
        const int kbase = kt * 64;
#pragma unroll
        for (int p = 0; p < 4; ++p) {
            const int idx = (tid + p * 256) * 8;
            { const int r = idx >> 7, c = idx & 127;   // Ks 64x128
              cp16_async(Kp + (size_t)(kbase + r) * DHEAD + c, &Ks[r][c]); }
            { const int r = idx >> 6, c = idx & 63;    // Vs 128x64
              cp16_async(Vp + (size_t)r * SEQ + kbase + c, &Vs[r][c]); }
        }
        wait_async_all();
        __syncthreads();

        // S = Q K^T   (16 x 64 per wave)
        v8f sacc[4];
#pragma unroll
        for (int nt = 0; nt < 4; ++nt) sacc[nt] = zero_v8f();
#pragma unroll
        for (int kk = 0; kk < 4; ++kk) {
#pragma unroll
            for (int nt = 0; nt < 4; ++nt) {
                v16bf bf = *(const v16bf*)&Ks[nt * 16 + l16][kk * 32 + lhi * 16];
                sacc[nt] = wmma_bf16(qf[kk], bf, sacc[nt]);
            }
        }

        // online softmax per row
#pragma unroll
        for (int i = 0; i < 8; ++i) {
            const int q = qt * 128 + wave * 16 + 8 * lhi + i;
            float rmax = -3.0e38f;
#pragma unroll
            for (int nt = 0; nt < 4; ++nt) {
                float s = sacc[nt][i] * SM_SCALE;
                if (kbase + nt * 16 + l16 > q) s = -10000.0f;   // causal mask (matches ref)
                sacc[nt][i] = s;
                rmax = fmaxf(rmax, s);
            }
            rmax = fmaxf(rmax, __shfl_xor(rmax, 1, 32));
            rmax = fmaxf(rmax, __shfl_xor(rmax, 2, 32));
            rmax = fmaxf(rmax, __shfl_xor(rmax, 4, 32));
            rmax = fmaxf(rmax, __shfl_xor(rmax, 8, 32));

            const float mnew = fmaxf(mrow[i], rmax);
            const float corr = __expf(mrow[i] - mnew);
            mrow[i] = mnew;

            float psum = 0.f;
            const int r = 8 * lhi + i;
#pragma unroll
            for (int nt = 0; nt < 4; ++nt) {
                float pv = __expf(sacc[nt][i] - mnew);
                Ps[wave][r][nt * 16 + l16] = (bf16)pv;
                psum += pv;
            }
            psum += __shfl_xor(psum, 1, 32);
            psum += __shfl_xor(psum, 2, 32);
            psum += __shfl_xor(psum, 4, 32);
            psum += __shfl_xor(psum, 8, 32);
            lrow[i] = lrow[i] * corr + psum;
#pragma unroll
            for (int dt = 0; dt < 8; ++dt) oacc[dt][i] *= corr;
        }

        // O += P V   (A = P from LDS, B = Vs [d][key])
#pragma unroll
        for (int kk = 0; kk < 2; ++kk) {
            v8bf lo = *(const v8bf*)&Ps[wave][l16][kk * 32 + lhi * 16];
            v8bf hi = *(const v8bf*)&Ps[wave][l16][kk * 32 + lhi * 16 + 8];
            v16bf pa = cat8(lo, hi);
#pragma unroll
            for (int dt = 0; dt < 8; ++dt) {
                v16bf bf = *(const v16bf*)&Vs[dt * 16 + l16][kk * 32 + lhi * 16];
                oacc[dt] = wmma_bf16(pa, bf, oacc[dt]);
            }
        }
        __syncthreads();   // all waves done reading Ks/Vs before next overwrite
    }

    // normalize + write ctx [B,S,E] bf16
#pragma unroll
    for (int i = 0; i < 8; ++i) {
        const float inv = 1.0f / lrow[i];
        const int s = qt * 128 + wave * 16 + 8 * lhi + i;
#pragma unroll
        for (int dt = 0; dt < 8; ++dt) {
            ctx[((size_t)(b * SEQ + s)) * EMB + h * DHEAD + dt * 16 + l16] =
                (bf16)(oacc[dt][i] * inv);
        }
    }
}

// ----------------------------------------------------------------
extern "C" void kernel_launch(void* const* d_in, const int* in_sizes, int n_in,
                              void* d_out, int out_size, void* d_ws, size_t ws_size,
                              hipStream_t stream) {
    const float* x      = (const float*)d_in[0];
    const float* wqkv_w = (const float*)d_in[1];
    const float* wqkv_b = (const float*)d_in[2];
    const float* wo_w   = (const float*)d_in[3];
    const float* wo_b   = (const float*)d_in[4];

    char* ws = (char*)d_ws;
    bf16* xbf    = (bf16*)(ws);                       // 8192*2048*2   = 33554432
    bf16* wqkvbf = (bf16*)(ws + 33554432ull);         // 6144*2048*2   = 25165824
    bf16* wobf   = (bf16*)(ws + 58720256ull);         // 2048*2048*2   =  8388608
    bf16* Qr     = (bf16*)(ws + 67108864ull);         // 33554432
    bf16* Kr     = (bf16*)(ws + 100663296ull);        // 33554432
    bf16* Vt     = (bf16*)(ws + 134217728ull);        // 33554432
    bf16* ctx    = (bf16*)(ws + 167772160ull);        // 33554432  (total ~201.3 MB)

    cvt_f32_bf16<<<4096, 256, 0, stream>>>(x,      xbf,    MROWS * EMB);
    cvt_f32_bf16<<<3072, 256, 0, stream>>>(wqkv_w, wqkvbf, 3 * EMB * EMB);
    cvt_f32_bf16<<<1024, 256, 0, stream>>>(wo_w,   wobf,   EMB * EMB);

    gemm_bf16<<<dim3(48, 64), 256, 0, stream>>>(xbf, wqkvbf, wqkv_b,
                                                3 * EMB, EMB, /*mode=*/0,
                                                Qr, Kr, Vt, nullptr);

    attn_flash<<<dim3(SEQ / 128, HEADS, BATCH), 256, 0, stream>>>(Qr, Kr, Vt, ctx);

    gemm_bf16<<<dim3(16, 64), 256, 0, stream>>>(ctx, wobf, wo_b,
                                                EMB, EMB, /*mode=*/1,
                                                nullptr, nullptr, nullptr,
                                                (float*)d_out);
}
